// ProteinTreeEncoder_90460601188797
// MI455X (gfx1250) — compile-verified, matrix operand order checked by reading, not run on hardware
//
#include <hip/hip_runtime.h>
#include <hip/hip_bf16.h>

typedef __attribute__((ext_vector_type(16))) __bf16 v16bf;
typedef __attribute__((ext_vector_type(8)))  float  v8f;

// ---------- helpers ----------
__device__ __forceinline__ unsigned short f2bf(float f) {
    unsigned u = __float_as_uint(f);
    unsigned r = (u + 0x7FFFu + ((u >> 16) & 1u)) >> 16;   // round-to-nearest-even
    return (unsigned short)r;
}
__device__ __forceinline__ float bf2f(unsigned b) {
    return __uint_as_float((b & 0xFFFFu) << 16);
}
__device__ __forceinline__ float sigm(float x) { return 1.0f / (1.0f + __expf(-x)); }

// k-index inside a 16x32 bf16 A (or 32x16 B) fragment for dword d, lane-half grp
__device__ __forceinline__ int kidx(int d, int grp) {
    return (d < 4) ? (d * 2 + grp * 8) : (16 + (d - 4) * 2 + grp * 8);
}

union Frag { uint4 q[2]; unsigned u[8]; v16bf v; };

__device__ __forceinline__ v8f wmma_bf16(v16bf a, v16bf b, v8f c) {
    return __builtin_amdgcn_wmma_f32_16x16x32_bf16(false, a, false, b,
                                                   (short)0, c, false, false);
}

// ---------- prep 1: leaf (V=21) h/c tables + symbol (S=50) gate-x table ----------
__global__ __launch_bounds__(256) void prep_tables(
    const float* __restrict__ emb, const float* __restrict__ sym_emb,
    const float* __restrict__ Wp, const float* __restrict__ bp,
    const float* __restrict__ Wi, const float* __restrict__ bi,
    const float* __restrict__ Wo, const float* __restrict__ bo,
    const float* __restrict__ Wu, const float* __restrict__ bu,
    const float* __restrict__ Wf, const float* __restrict__ bf_,
    unsigned short* __restrict__ leaf_h, unsigned short* __restrict__ leaf_c,
    float* __restrict__ sym_gx)
{
    __shared__ float xl[21 * 128];
    __shared__ float xs[50 * 128];
    int tid = threadIdx.x;
    for (int i = tid; i < 21 * 128; i += 256) {
        int t = i >> 7, h = i & 127;
        float acc = bp[h];
        if (t != 0)                       // emb.at[0].set(0)
            for (int e = 0; e < 64; ++e) acc += emb[t * 64 + e] * Wp[h * 64 + e];
        xl[i] = acc;
    }
    for (int i = tid; i < 50 * 128; i += 256) {
        int s = i >> 7, h = i & 127;
        float acc = bp[h];
        for (int e = 0; e < 64; ++e) acc += sym_emb[s * 64 + e] * Wp[h * 64 + e];
        xs[i] = acc;
    }
    __syncthreads();
    for (int i = tid; i < 21 * 128; i += 256) {
        int t = i >> 7, h = i & 127;
        float ai = bi[h], ao = bo[h], au = bu[h];
        for (int k = 0; k < 128; ++k) {
            float x = xl[t * 128 + k];
            ai += x * Wi[h * 128 + k];
            ao += x * Wo[h * 128 + k];
            au += x * Wu[h * 128 + k];
        }
        float gi = sigm(ai), go = sigm(ao), gu = tanhf(au);
        float c = gi * gu, hh = go * tanhf(c);
        leaf_h[i] = f2bf(hh);
        leaf_c[i] = f2bf(c);
    }
    for (int i = tid; i < 50 * 512; i += 256) {
        int s = i >> 9, r = i & 511, g = r >> 7, h = r & 127;
        const float* W  = (g == 0) ? Wi : (g == 1) ? Wo : (g == 2) ? Wu : Wf;
        const float* bb = (g == 0) ? bi : (g == 1) ? bo : (g == 2) ? bu : bf_;
        float acc = bb[h];
        for (int k = 0; k < 128; ++k) acc += xs[s * 128 + k] * W[h * 128 + k];
        sym_gx[i] = acc;
    }
}

// ---------- prep 2: swizzle U matrices into bf16 WMMA B-fragments ----------
// frag 0..95: [Ui|Uo|Uu]^T, t=frag/4 (n-tile 0..23), s=frag%4 (k-step)
// frag 96..127: Uf^T, t=(frag-96)/4 (n-tile 0..7), s=%4
__global__ __launch_bounds__(256) void prep_frags(
    const float* __restrict__ Ui, const float* __restrict__ Uo,
    const float* __restrict__ Uu, const float* __restrict__ Uf,
    unsigned* __restrict__ frags)
{
    int gid = blockIdx.x * 256 + threadIdx.x;
    if (gid >= 128 * 32) return;
    int frag = gid >> 5, lane = gid & 31;
    int grp = lane >> 4, nl = lane & 15;
    const float* U; int nrow, s;
    if (frag < 96) {
        int t = frag >> 2; s = frag & 3;
        int n = t * 16 + nl, g = n >> 7, w = n & 127;
        U = (g == 0) ? Ui : (g == 1) ? Uo : Uu;
        nrow = w;
    } else {
        int f2 = frag - 96; int t = f2 >> 2; s = f2 & 3;
        U = Uf; nrow = t * 16 + nl;
    }
#pragma unroll
    for (int d = 0; d < 8; ++d) {
        int k = s * 32 + kidx(d, grp);
        unsigned lo = f2bf(U[nrow * 128 + k]);
        unsigned hi = f2bf(U[nrow * 128 + k + 1]);
        frags[frag * 256 + lane * 8 + d] = lo | (hi << 16);
    }
}

// ---------- leaf fill: pure bf16 gather from 21-entry table ----------
__global__ __launch_bounds__(256) void leaf_fill(
    const int* __restrict__ tokens,
    const unsigned long long* __restrict__ leaf_h64,
    const unsigned long long* __restrict__ leaf_c64,
    unsigned long long* __restrict__ h0, unsigned long long* __restrict__ c0)
{
    int idx = blockIdx.x * 256 + threadIdx.x;      // 64*4096*32 chunks of 4 cols
    int node = idx >> 5, chunk = idx & 31;
    int tok = tokens[node];
    h0[(size_t)node * 32 + chunk] = leaf_h64[tok * 32 + chunk];
    c0[(size_t)node * 32 + chunk] = leaf_c64[tok * 32 + chunk];
}

// ---------- one tree merge level (the WMMA workhorse) ----------
// m = merges/batch = 1<<mlog ; input has 2m nodes/batch, output m nodes/batch
// 256 threads = 8 waves/WG sharing one 128KB LDS fragment cache
// -> 2 WGs/WGP (320KB LDS) = 4 waves per SIMD32 for latency hiding.
__global__ __launch_bounds__(256) void tree_level(
    const unsigned short* __restrict__ h_in, const unsigned short* __restrict__ c_in,
    unsigned short* __restrict__ h_out, unsigned short* __restrict__ c_out,
    const unsigned* __restrict__ frags, const float* __restrict__ sym_gx,
    const int* __restrict__ symbols, int mlog, int off)
{
    extern __shared__ unsigned lds_u[];            // 128 KB of B-fragments
    {   // stage all 128 fragments into LDS (coalesced b128)
        const uint4* src = (const uint4*)frags;
        uint4* dst = (uint4*)lds_u;
        for (int i = threadIdx.x; i < 8192; i += 256) dst[i] = src[i];
    }
    __syncthreads();

    const int m     = 1 << mlog;
    const int lane  = threadIdx.x & 31;
    const int wid   = threadIdx.x >> 5;            // 0..7
    const int grp   = lane >> 4;
    const int nTiles = (64 * m) >> 4;

    for (int tile = blockIdx.x * 8 + wid; tile < nTiles; tile += gridDim.x * 8) {
        // ---- A fragments for hl / hr / ht=hl+hr (K=128 -> 4 k-steps) ----
        // Row-major bf16 rows load directly as 2x b128 per fragment (ISA layout).
        int r  = tile * 16 + (lane & 15);
        int b  = r >> mlog, jj = r & (m - 1);
        size_t rowL = ((size_t)((b << (mlog + 1)) + (jj << 1))) << 7;
        size_t rowR = rowL + 128;
        Frag fhl[4], fhr[4], fht[4];
#pragma unroll
        for (int s = 0; s < 4; ++s) {
            size_t base = (size_t)s * 32 + grp * 8;
            fhl[s].q[0] = *(const uint4*)(h_in + rowL + base);
            fhl[s].q[1] = *(const uint4*)(h_in + rowL + base + 16);
            fhr[s].q[0] = *(const uint4*)(h_in + rowR + base);
            fhr[s].q[1] = *(const uint4*)(h_in + rowR + base + 16);
            fht[s].v    = fhl[s].v + fhr[s].v;     // packed bf16 add
        }
        // ---- per-node metadata, hoisted out of the column-block loop ----
        int    symv[8];
        size_t crowv[8], orowv[8];
#pragma unroll
        for (int v = 0; v < 8; ++v) {
            int g  = tile * 16 + v + (grp << 3);
            int bb = g >> mlog, j2 = g & (m - 1);
            symv[v]  = symbols[bb * 4095 + off + j2];
            crowv[v] = ((size_t)((bb << (mlog + 1)) + (j2 << 1))) << 7;
            orowv[v] = ((size_t)g) << 7;
        }
        // ---- 8 column blocks of 16; 5 gate tiles each; 4 k-steps -> 160 WMMA
        for (int nb = 0; nb < 8; ++nb) {
            v8f aI = {0,0,0,0,0,0,0,0}, aO = {0,0,0,0,0,0,0,0}, aU = {0,0,0,0,0,0,0,0};
            v8f aFL = {0,0,0,0,0,0,0,0}, aFR = {0,0,0,0,0,0,0,0};
#pragma unroll
            for (int s = 0; s < 4; ++s) {
                Frag bi_, bo_, bu_, bf_;
                const uint4* p;
                p = (const uint4*)(lds_u + (((0 * 8 + nb) * 4 + s) * 256) + lane * 8);
                bi_.q[0] = p[0]; bi_.q[1] = p[1];
                p = (const uint4*)(lds_u + (((1 * 8 + nb) * 4 + s) * 256) + lane * 8);
                bo_.q[0] = p[0]; bo_.q[1] = p[1];
                p = (const uint4*)(lds_u + (((2 * 8 + nb) * 4 + s) * 256) + lane * 8);
                bu_.q[0] = p[0]; bu_.q[1] = p[1];
                p = (const uint4*)(lds_u + ((96 + nb * 4 + s) * 256) + lane * 8);
                bf_.q[0] = p[0]; bf_.q[1] = p[1];
                aI  = wmma_bf16(fht[s].v, bi_.v, aI);
                aO  = wmma_bf16(fht[s].v, bo_.v, aO);
                aU  = wmma_bf16(fht[s].v, bu_.v, aU);
                aFL = wmma_bf16(fhl[s].v, bf_.v, aFL);
                aFR = wmma_bf16(fhr[s].v, bf_.v, aFR);
            }
            // ---- gated elementwise combine; C/D layout: lane->N, vgpr->M ----
            int col = nb * 16 + (lane & 15);
#pragma unroll
            for (int v = 0; v < 8; ++v) {
                const float* gx = sym_gx + symv[v] * 512;
                float cl = bf2f(c_in[crowv[v] + col]);
                float cr = bf2f(c_in[crowv[v] + 128 + col]);
                float gi = sigm(aI[v] + gx[col]);
                float go = sigm(aO[v] + gx[128 + col]);
                float gu = tanhf(aU[v] + gx[256 + col]);
                float fx = gx[384 + col];
                float fl = sigm(aFL[v] + fx);
                float fr = sigm(aFR[v] + fx);
                float cn = gi * gu + fl * cl + fr * cr;
                float hn = go * tanhf(cn);
                h_out[orowv[v] + col] = f2bf(hn);
                c_out[orowv[v] + col] = f2bf(cn);
            }
        }
    }
}

// ---------- final projection: (64,128) @ Wout^T + bout ----------
__global__ __launch_bounds__(256) void out_proj(
    const unsigned short* __restrict__ rootH,
    const float* __restrict__ Wout, const float* __restrict__ bout,
    float* __restrict__ out)
{
    int idx = blockIdx.x * 256 + threadIdx.x;
    if (idx >= 64 * 128) return;
    int b = idx >> 7, n = idx & 127;
    float acc = bout[n];
    for (int k = 0; k < 128; ++k)
        acc += bf2f(rootH[b * 128 + k]) * Wout[n * 128 + k];
    out[idx] = acc;
}

// ---------- host ----------
static inline size_t align256(size_t x) { return (x + 255) & ~(size_t)255; }

extern "C" void kernel_launch(void* const* d_in, const int* in_sizes, int n_in_,
                              void* d_out, int out_size, void* d_ws, size_t ws_size,
                              hipStream_t stream) {
    const int*   tokens  = (const int*)  d_in[0];
    const int*   symbols = (const int*)  d_in[1];
    const float* emb     = (const float*)d_in[2];
    const float* sym_emb = (const float*)d_in[3];
    const float* Wp = (const float*)d_in[4],  *bp  = (const float*)d_in[5];
    const float* Wi = (const float*)d_in[6],  *bi  = (const float*)d_in[7];
    const float* Ui = (const float*)d_in[8];
    const float* Wf = (const float*)d_in[9],  *bf_ = (const float*)d_in[10];
    const float* Uf = (const float*)d_in[11];
    const float* Wo = (const float*)d_in[12], *bo  = (const float*)d_in[13];
    const float* Uo = (const float*)d_in[14];
    const float* Wu = (const float*)d_in[15], *bu  = (const float*)d_in[16];
    const float* Uu = (const float*)d_in[17];
    const float* Wout = (const float*)d_in[18], *bout = (const float*)d_in[19];

    char* ws = (char*)d_ws;
    size_t o = 0;
    unsigned short* leaf_h = (unsigned short*)(ws + o); o = align256(o + 21 * 128 * 2);
    unsigned short* leaf_c = (unsigned short*)(ws + o); o = align256(o + 21 * 128 * 2);
    float*          sym_gx = (float*)        (ws + o); o = align256(o + 50 * 512 * 4);
    unsigned*       frags  = (unsigned*)     (ws + o); o = align256(o + 128 * 1024);
    unsigned short* hA = (unsigned short*)(ws + o); o = align256(o + (size_t)64 * 4096 * 128 * 2);
    unsigned short* cA = (unsigned short*)(ws + o); o = align256(o + (size_t)64 * 4096 * 128 * 2);
    unsigned short* hB = (unsigned short*)(ws + o); o = align256(o + (size_t)64 * 2048 * 128 * 2);
    unsigned short* cB = (unsigned short*)(ws + o); o = align256(o + (size_t)64 * 2048 * 128 * 2);
    (void)ws_size; (void)in_sizes; (void)n_in_;

    hipFuncSetAttribute((const void*)tree_level,
                        hipFuncAttributeMaxDynamicSharedMemorySize, 131072);

    prep_tables<<<1, 256, 0, stream>>>(emb, sym_emb, Wp, bp, Wi, bi, Wo, bo,
                                       Wu, bu, Wf, bf_, leaf_h, leaf_c, sym_gx);
    prep_frags<<<16, 256, 0, stream>>>(Ui, Uo, Uu, Uf, frags);
    leaf_fill<<<(64 * 4096 * 32) / 256, 256, 0, stream>>>(
        tokens, (const unsigned long long*)leaf_h, (const unsigned long long*)leaf_c,
        (unsigned long long*)hA, (unsigned long long*)cA);

    const unsigned short* hcur = hA;
    const unsigned short* ccur = cA;
    int offsym = 0;
    bool useB = true;
    for (int mlog = 11; mlog >= 0; --mlog) {       // m = 2048 ... 1
        int m = 1 << mlog;
        unsigned short* ho = useB ? hB : hA;
        unsigned short* co = useB ? cB : cA;
        int nTiles = (64 * m) / 16;
        int wgs = (nTiles + 7) / 8;
        if (wgs > 1024) wgs = 1024;
        tree_level<<<wgs, 256, 131072, stream>>>(hcur, ccur, ho, co,
                                                 frags, sym_gx, symbols,
                                                 mlog, offsym);
        hcur = ho; ccur = co;
        offsym += m; useB = !useB;
    }

    out_proj<<<(64 * 128 + 255) / 256, 256, 0, stream>>>(hcur, Wout, bout, (float*)d_out);
}
